// GRUNet_31336081391696
// MI455X (gfx1250) — compile-verified, matrix operand order checked by reading, not run on hardware
//
#include <hip/hip_runtime.h>
#include <hip/hip_bf16.h>

// ---------------------------------------------------------------------------
// GRU on gfx1250 (MI455X): persistent bf16-WMMA kernel, TDM-staged
// LDS-resident weights, device-scope grid barrier between phases.
//   I=512, H=1024, B=64, S=512, O=2
// ---------------------------------------------------------------------------

typedef __attribute__((ext_vector_type(16))) __bf16 v16bf;
typedef __attribute__((ext_vector_type(8)))  __bf16 v8bf;
typedef __attribute__((ext_vector_type(8)))  float  v8f;
typedef unsigned int v4u __attribute__((ext_vector_type(4)));
typedef int          v8i __attribute__((ext_vector_type(8)));
typedef int          v4i __attribute__((ext_vector_type(4)));

#define I_DIM 512
#define H_DIM 1024
#define B_DIM 64
#define S_DIM 512
#define KCAT  (I_DIM + H_DIM)   // 1536
#define NBLK  96                // 32 z + 32 r + 32 cand blocks

// LDS weight layout matches TDM padding: 16B pad after every 1024B stored.
// Row = 3072B = 3 intervals -> row stride 3120B (780 dwords == 12 mod 64
// banks: conflict-free ds_load_b128 within each half-wave).
#define KPADE 1560              // row stride in bf16 elements (3120 B)
#define SMEM_BYTES (32 * KPADE * 2)   // 99840 B

// Element offset of K index k inside a padded LDS row (pads at 512-elem marks;
// 32-aligned K tiles never straddle one).
#define LDS_K(k) ((k) + (((k) >> 9) << 3))

// ---------------------------------------------------------------------------
// Prep kernels (one-time).
// ---------------------------------------------------------------------------
__global__ void conv_x_kernel(const float* __restrict__ x,
                              __bf16* __restrict__ xb, int n) {
  int i = (blockIdx.x * blockDim.x + threadIdx.x) * 4;
  if (i + 3 < n) {
    float4 v = *(const float4*)(x + i);
    xb[i + 0] = (__bf16)v.x;
    xb[i + 1] = (__bf16)v.y;
    xb[i + 2] = (__bf16)v.z;
    xb[i + 3] = (__bf16)v.w;
  }
}

// W (KCAT x H) fp32 -> Wt (H x KCAT) bf16 so the WMMA B-operand (and the TDM
// tile source) reads contiguous K.
__global__ void conv_wt_kernel(const float* __restrict__ Wz,
                               const float* __restrict__ Wr,
                               const float* __restrict__ Wh,
                               __bf16* __restrict__ Wzt,
                               __bf16* __restrict__ Wrt,
                               __bf16* __restrict__ Wht) {
  const float* Ws[3] = {Wz, Wr, Wh};
  __bf16*      Ts[3] = {Wzt, Wrt, Wht};
  const float* W = Ws[blockIdx.y];
  __bf16*      T = Ts[blockIdx.y];
  int idx = blockIdx.x * blockDim.x + threadIdx.x;   // over KCAT*H
  if (idx < KCAT * H_DIM) {
    int k = idx / H_DIM;
    int n = idx % H_DIM;
    T[(size_t)n * KCAT + k] = (__bf16)W[idx];
  }
}

__global__ void init_h_kernel(const float* __restrict__ h,
                              float* __restrict__ hf,
                              __bf16* __restrict__ hb, int n) {
  int i = blockIdx.x * blockDim.x + threadIdx.x;
  if (i < n) {
    float v = h[i];
    hf[i] = v;
    hb[i] = (__bf16)v;
  }
}

__global__ void init_sync_kernel(unsigned* __restrict__ cnt,
                                 unsigned* __restrict__ gen) {
  if (threadIdx.x == 0) { *cnt = 0u; *gen = 0u; }
}

// ---------------------------------------------------------------------------
// Grid-wide sense barrier (all NBLK blocks co-resident).
// ---------------------------------------------------------------------------
__device__ __forceinline__ void grid_barrier(unsigned* cnt, unsigned* gen) {
  __builtin_amdgcn_fence(__ATOMIC_RELEASE, "agent");
  __syncthreads();
  if (threadIdx.x == 0) {
    unsigned g = __hip_atomic_load(gen, __ATOMIC_RELAXED,
                                   __HIP_MEMORY_SCOPE_AGENT);
    unsigned arrived = __hip_atomic_fetch_add(cnt, 1u, __ATOMIC_ACQ_REL,
                                              __HIP_MEMORY_SCOPE_AGENT);
    if (arrived == NBLK - 1) {
      __hip_atomic_store(cnt, 0u, __ATOMIC_RELAXED, __HIP_MEMORY_SCOPE_AGENT);
      __hip_atomic_store(gen, g + 1u, __ATOMIC_RELEASE,
                         __HIP_MEMORY_SCOPE_AGENT);
    } else {
      while (__hip_atomic_load(gen, __ATOMIC_ACQUIRE,
                               __HIP_MEMORY_SCOPE_AGENT) == g) {
        __builtin_amdgcn_s_sleep(1);
      }
    }
  }
  __syncthreads();
  __builtin_amdgcn_fence(__ATOMIC_ACQUIRE, "agent");
}

// Assemble a 32B WMMA operand from two 16B-aligned LDS halves.
__device__ __forceinline__ v16bf lds_b16(const __bf16* p) {
  v8bf lo = *(const v8bf*)(p);
  v8bf hi = *(const v8bf*)(p + 8);
  return __builtin_shufflevector(lo, hi, 0, 1, 2, 3, 4, 5, 6, 7, 8, 9, 10, 11,
                                 12, 13, 14, 15);
}

__device__ __forceinline__ v8f wmma_bf16(v16bf a, v16bf b, v8f c) {
  return __builtin_amdgcn_wmma_f32_16x16x32_bf16(false, a, false, b, (short)0,
                                                 c, false, false);
}

// ---------------------------------------------------------------------------
// Stage this block's 32 x 1536 bf16 weight slice into padded LDS.
// Preferred path: one Tensor Data Mover descriptor; HW applies the 16B pad
// every 1024B (pad_interval=7 -> 256 dwords, pad_amount=3 -> 4 dwords).
// This toolchain's builtin is the 6-operand form (g0..g3 + extra group + cpol).
// ---------------------------------------------------------------------------
__device__ __forceinline__ void stage_weights(__bf16* smem,
                                              const __bf16* __restrict__ Wt,
                                              int col0, int wave) {
#if defined(__has_builtin) && __has_builtin(__builtin_amdgcn_tensor_load_to_lds)
  if (wave == 0) {
    const unsigned long long ga =
        (unsigned long long)(size_t)(Wt + (size_t)col0 * KCAT);
    const unsigned lds_off = (unsigned)(size_t)smem;  // aperture low 32 = offset
    // D# group 0: count=1 | lds_addr | global_addr(57b) | type=2
    v4u g0;
    g0[0] = 1u;
    g0[1] = lds_off;
    g0[2] = (unsigned)(ga & 0xffffffffu);
    g0[3] = (unsigned)((ga >> 32) & 0x01ffffffu) | 0x80000000u;
    // D# group 1: data_size=2B, pad_enable, pad_interval=7 (1024B),
    // pad_amount=3 (4 dwords); tensor_dim0=1536, tensor_dim1=32,
    // tile_dim0=1536, tile_dim1=32, tensor_dim0_stride=1536.
    v8i g1;
    g1[0] = (int)((1u << 16) | (1u << 20) | (7u << 22) | (3u << 25));
    g1[1] = (int)(1536u << 16);   // tensor_dim0 lo16 in bits 63:48
    g1[2] = (int)(32u << 16);     // tensor_dim1 lo16 in bits 95:80
    g1[3] = (int)(1536u << 16);   // tile_dim0 in bits 127:112
    g1[4] = 32;                   // tile_dim1 in bits 143:128
    g1[5] = 1536;                 // tensor_dim0_stride lo32
    g1[6] = 0;
    g1[7] = 0;
    v4i gz4 = {0, 0, 0, 0};            // 2D tile: groups 2/3 zero
    v8i gz8 = {0, 0, 0, 0, 0, 0, 0, 0};
    __builtin_amdgcn_tensor_load_to_lds(g0, g1, gz4, gz4, gz8, 0);
    __builtin_amdgcn_s_wait_tensorcnt(0);
  }
#else
  // Fallback: cooperative copy writing the identical padded layout.
  for (int c = threadIdx.x; c < 32 * (KCAT / 8); c += 256) {
    int row = c / (KCAT / 8);
    int k8  = (c % (KCAT / 8)) * 8;
    *(v8bf*)&smem[(size_t)row * KPADE + LDS_K(k8)] =
        *(const v8bf*)&Wt[(size_t)(col0 + row) * KCAT + k8];
  }
#endif
  __syncthreads();
}

// ---------------------------------------------------------------------------
// Persistent GRU kernel. 96 blocks x 256 threads (8 waves; 4 M-tiles x 2
// N-tiles of 16x16 WMMA). Block roles:
//   blk  0..31 : z gate for cols [32b, 32b+32)
//   blk 32..63 : r gate
//   blk 64..95 : candidate (x-part in phase 1, r*h-part in phase 2)
// Weight slice TDM-staged once into LDS, reused for all 512 steps.
// ---------------------------------------------------------------------------
__global__ __launch_bounds__(256)
void gru_persistent_kernel(const __bf16* __restrict__ xb,
                           const __bf16* __restrict__ Wzt,
                           const __bf16* __restrict__ Wrt,
                           const __bf16* __restrict__ Wht,
                           const float* __restrict__ bz,
                           const float* __restrict__ br,
                           const float* __restrict__ bh,
                           float* __restrict__ hf0, float* __restrict__ hf1,
                           __bf16* __restrict__ hbf0,
                           __bf16* __restrict__ hbf1,
                           float* __restrict__ zbuf,
                           __bf16* __restrict__ rhbf,
                           unsigned* __restrict__ cnt,
                           unsigned* __restrict__ gen) {
  extern __shared__ __bf16 smem[];

  const int blk  = blockIdx.x;
  const int role = blk >> 5;          // 0=z, 1=r, 2=cand
  const int col0 = (blk & 31) * 32;
  const int lane = threadIdx.x & 31;
  const int wave = threadIdx.x >> 5;
  const int m0   = (wave & 3) * 16;
  const int nloc = (wave >> 2) * 16;  // 0 or 16 inside the 32-col slice
  const int n0   = col0 + nloc;

  const __bf16* Wt = (role == 0) ? Wzt : (role == 1) ? Wrt : Wht;
  stage_weights(smem, Wt, col0, wave);

  const int r    = lane & 15;
  const int koff = lane & 16;   // 0 or 16

  const __bf16* bsm   = smem + (size_t)(nloc + r) * KPADE + koff;
  const int     nn    = n0 + (lane & 15);
  const int     mbase = m0 + ((lane >> 4) << 3);
  const float   bval  = (role == 0) ? bz[nn] : (role == 1) ? br[nn] : bh[nn];

  for (int t = 0; t < S_DIM; ++t) {
    const float*  hfc = (t & 1) ? hf1 : hf0;
    float*        hfn = (t & 1) ? hf0 : hf1;
    const __bf16* hbc = (t & 1) ? hbf1 : hbf0;
    __bf16*       hbn = (t & 1) ? hbf0 : hbf1;

    const __bf16* axp = xb + ((size_t)(m0 + r) * S_DIM + t) * I_DIM + koff;

    v8f c = {};
    // ---- x-part of the concat GEMM (all roles); K<512 -> no LDS pad ----
    #pragma unroll 4
    for (int k0 = 0; k0 < I_DIM; k0 += 32) {
      __builtin_prefetch(axp + k0 + 128, 0, 0);
      v16bf a = *(const v16bf*)(axp + k0);
      v16bf b = lds_b16(bsm + k0);
      c = wmma_bf16(a, b, c);
    }

    if (role < 2) {
      // ---- h-part; epilogue: sigmoid -> z or r*h ----
      const __bf16* ahp = hbc + (size_t)(m0 + r) * H_DIM + koff;
      #pragma unroll 4
      for (int k0 = 0; k0 < H_DIM; k0 += 32) {
        v16bf a = *(const v16bf*)(ahp + k0);
        v16bf b = lds_b16(bsm + LDS_K(I_DIM + k0));
        c = wmma_bf16(a, b, c);
      }
      #pragma unroll
      for (int v = 0; v < 8; ++v) {
        const size_t idx = (size_t)(mbase + v) * H_DIM + nn;
        const float  g   = 1.0f / (1.0f + __expf(-(c[v] + bval)));
        if (role == 0) zbuf[idx] = g;
        else           rhbf[idx] = (__bf16)(g * hfc[idx]);
      }
    }

    grid_barrier(cnt, gen);   // r (and z) now globally visible

    if (role == 2) {
      // ---- finish candidate over r*h; fused state update ----
      const __bf16* arp = rhbf + (size_t)(m0 + r) * H_DIM + koff;
      #pragma unroll 4
      for (int k0 = 0; k0 < H_DIM; k0 += 32) {
        v16bf a = *(const v16bf*)(arp + k0);
        v16bf b = lds_b16(bsm + LDS_K(I_DIM + k0));
        c = wmma_bf16(a, b, c);
      }
      #pragma unroll
      for (int v = 0; v < 8; ++v) {
        const size_t idx  = (size_t)(mbase + v) * H_DIM + nn;
        const float  hcv  = tanhf(c[v] + bval);
        const float  z    = zbuf[idx];
        const float  hold = hfc[idx];
        const float  hnew = (1.0f - z) * hold + z * hcv;
        hfn[idx] = hnew;
        hbn[idx] = (__bf16)hnew;
      }
    }

    grid_barrier(cnt, gen);   // h' globally visible for step t+1
  }
}

// ---------------------------------------------------------------------------
// Final: out = log_softmax(relu(h) @ Wfc + bfc).  One wave per batch row.
// ---------------------------------------------------------------------------
__global__ __launch_bounds__(32)
void gru_final_kernel(const float* __restrict__ hf,
                      const float* __restrict__ Wfc,
                      const float* __restrict__ bfc,
                      float* __restrict__ out) {
  const int row  = blockIdx.x;
  const int lane = threadIdx.x;
  float s0 = 0.0f, s1 = 0.0f;
  for (int k = lane; k < H_DIM; k += 32) {
    const float a = fmaxf(hf[(size_t)row * H_DIM + k], 0.0f);
    s0 += a * Wfc[k * 2 + 0];
    s1 += a * Wfc[k * 2 + 1];
  }
  #pragma unroll
  for (int off = 16; off > 0; off >>= 1) {
    s0 += __shfl_xor(s0, off, 32);
    s1 += __shfl_xor(s1, off, 32);
  }
  if (lane == 0) {
    const float l0 = s0 + bfc[0];
    const float l1 = s1 + bfc[1];
    const float mx = fmaxf(l0, l1);
    const float lse = mx + __logf(__expf(l0 - mx) + __expf(l1 - mx));
    out[row * 2 + 0] = l0 - lse;
    out[row * 2 + 1] = l1 - lse;
  }
}

// ---------------------------------------------------------------------------
// Host launcher: 6 launches total.
// ---------------------------------------------------------------------------
extern "C" void kernel_launch(void* const* d_in, const int* in_sizes, int n_in,
                              void* d_out, int out_size, void* d_ws,
                              size_t ws_size, hipStream_t stream) {
  const float* x   = (const float*)d_in[0];
  const float* h0  = (const float*)d_in[1];
  const float* Wz  = (const float*)d_in[2];
  const float* bz  = (const float*)d_in[3];
  const float* Wr  = (const float*)d_in[4];
  const float* br  = (const float*)d_in[5];
  const float* Wh  = (const float*)d_in[6];
  const float* bh  = (const float*)d_in[7];
  const float* Wfc = (const float*)d_in[8];
  const float* bfc = (const float*)d_in[9];
  float* out = (float*)d_out;

  // ---- workspace carve (256B aligned) ----
  char*  ws  = (char*)d_ws;
  size_t off = 0;
  auto carve = [&](size_t bytes) -> void* {
    void* p = ws + off;
    off = (off + bytes + 255) & ~(size_t)255;
    return p;
  };
  const size_t nX = (size_t)B_DIM * S_DIM * I_DIM;  // 16.7M
  const size_t nH = (size_t)B_DIM * H_DIM;          // 64K
  const size_t nW = (size_t)KCAT * H_DIM;           // 1.57M

  __bf16*   xb   = (__bf16*)carve(nX * sizeof(__bf16));
  __bf16*   Wzt  = (__bf16*)carve(nW * sizeof(__bf16));
  __bf16*   Wrt  = (__bf16*)carve(nW * sizeof(__bf16));
  __bf16*   Wht  = (__bf16*)carve(nW * sizeof(__bf16));
  float*    zbuf = (float*)carve(nH * sizeof(float));
  __bf16*   rhbf = (__bf16*)carve(nH * sizeof(__bf16));
  float*    hf0  = (float*)carve(nH * sizeof(float));
  float*    hf1  = (float*)carve(nH * sizeof(float));
  __bf16*   hbf0 = (__bf16*)carve(nH * sizeof(__bf16));
  __bf16*   hbf1 = (__bf16*)carve(nH * sizeof(__bf16));
  unsigned* sync = (unsigned*)carve(2 * sizeof(unsigned));

  // ---- one-time prep ----
  conv_x_kernel<<<(int)(nX / 4 / 256), 256, 0, stream>>>(x, xb, (int)nX);
  conv_wt_kernel<<<dim3((nW + 255) / 256, 3), 256, 0, stream>>>(Wz, Wr, Wh, Wzt,
                                                                Wrt, Wht);
  init_h_kernel<<<(int)((nH + 255) / 256), 256, 0, stream>>>(h0, hf0, hbf0,
                                                             (int)nH);
  init_sync_kernel<<<1, 32, 0, stream>>>(sync, sync + 1);

  // ---- persistent scan: weights live in LDS for all 512 steps ----
  (void)hipFuncSetAttribute((const void*)gru_persistent_kernel,
                            hipFuncAttributeMaxDynamicSharedMemorySize,
                            SMEM_BYTES);

  gru_persistent_kernel<<<NBLK, 256, SMEM_BYTES, stream>>>(
      xb, Wzt, Wrt, Wht, bz, br, bh, hf0, hf1, hbf0, hbf1, zbuf, rhbf, sync,
      sync + 1);

  // S=512 even -> final state in buffer 0.
  gru_final_kernel<<<B_DIM, 32, 0, stream>>>(hf0, Wfc, bfc, out);
}
